// AdvancedEdgeConvLayer_31782757990847
// MI455X (gfx1250) — matrix operand order repaired and checked by reading, not compiled, not run on hardware
//
#include <hip/hip_runtime.h>

// ---------------------------------------------------------------------------
// Problem constants (match reference)
// ---------------------------------------------------------------------------
#define N_NODES    100000
#define NODE_DIM   128
#define EDGE_DIM   64
#define HIDDEN_DIM 256
#define N_EDGES    640000
#define IN_DIM     320            // 2*NODE_DIM + EDGE_DIM
#define TILE_E     64             // edges per workgroup (640000 % 64 == 0)
#define MS         328            // msg LDS row stride (bf16), 16B-aligned rows
#define HS         264            // h   LDS row stride (bf16), 16B-aligned rows
#define NBLOCKS    (N_EDGES / TILE_E)

typedef __attribute__((ext_vector_type(16))) __bf16 v16bf;
typedef __attribute__((ext_vector_type(8)))  __bf16 v8bf;
typedef __attribute__((ext_vector_type(4)))  __bf16 v4bf;
typedef __attribute__((ext_vector_type(8)))  float  v8f;

__device__ __forceinline__ v16bf frag_cat(v8bf lo, v8bf hi) {
  return __builtin_shufflevector(lo, hi, 0,1,2,3,4,5,6,7,8,9,10,11,12,13,14,15);
}

// A fragment (16x32 bf16, ISA 7.12.2): lanes 0-15 hold M=lane with K chunks
// {kk+0..7, kk+16..23}; lanes 16-31 hold M=lane-16 with {kk+8..15, kk+24..31}.
__device__ __forceinline__ v16bf load_frag_a_lds(const __bf16* tile, int stride,
                                                 int mbase, int kk, int lane) {
  const int m  = mbase + (lane & 15);
  const int c0 = kk + ((lane >> 4) << 3);          // +0 or +8
  const __bf16* p = tile + m * stride + c0;
  v8bf lo = *(const v8bf*)p;                        // ds_load_b128
  v8bf hi = *(const v8bf*)(p + 16);                 // ds_load_b128
  return frag_cat(lo, hi);
}

// B fragment (32x16 bf16): lanes 0-15 hold N=lane, K={kk+0..7, kk+16..23};
// lanes 16-31 hold N=lane-16, K={kk+8..15, kk+24..31}.  Weights stored K-contiguous
// (transposed) so this is two global_load_b128 per lane.
__device__ __forceinline__ v16bf load_frag_b_glb(const __bf16* WT, int ldk,
                                                 int nbase, int kk, int lane) {
  const int n  = nbase + (lane & 15);
  const int k0 = kk + ((lane >> 4) << 3);          // +0 or +8
  const __bf16* p = WT + (size_t)n * ldk + k0;
  v8bf lo = *(const v8bf*)p;
  v8bf hi = *(const v8bf*)(p + 16);
  return frag_cat(lo, hi);
}

#define WMMA_BF16(a, b, c) \
  __builtin_amdgcn_wmma_f32_16x16x32_bf16(false, (a), false, (b), (short)0, (c), false, false)

// ---------------------------------------------------------------------------
// Weight prep: fp32 row-major -> bf16 K-contiguous (transposed) in workspace
// ---------------------------------------------------------------------------
__global__ void edgeconv_prep_weights(const float* __restrict__ W1,   // [320][256]
                                      const float* __restrict__ W2,   // [256][128]
                                      unsigned short* __restrict__ w1t_u,
                                      unsigned short* __restrict__ w2t_u) {
  __bf16* W1T = (__bf16*)w1t_u;                    // [256][320]
  __bf16* W2T = (__bf16*)w2t_u;                    // [128][256]
  const int idx    = blockIdx.x * blockDim.x + threadIdx.x;
  const int stride = gridDim.x * blockDim.x;
  for (int i = idx; i < IN_DIM * HIDDEN_DIM; i += stride) {
    int k = i / HIDDEN_DIM, n = i - k * HIDDEN_DIM;
    W1T[n * IN_DIM + k] = (__bf16)W1[i];
  }
  for (int i = idx; i < HIDDEN_DIM * NODE_DIM; i += stride) {
    int k = i / NODE_DIM, n = i - k * NODE_DIM;
    W2T[n * HIDDEN_DIM + k] = (__bf16)W2[i];
  }
}

// ---------------------------------------------------------------------------
// Zero-init output (segment_sum semantics; nodes with no edges -> 0)
// ---------------------------------------------------------------------------
__global__ void edgeconv_zero_out(float4* __restrict__ out4, int n4) {
  int idx = blockIdx.x * blockDim.x + threadIdx.x;
  if (idx < n4) out4[idx] = make_float4(0.f, 0.f, 0.f, 0.f);
}

// ---------------------------------------------------------------------------
// Main fused kernel: gather -> (320->256 relu) -> (256->128) -> scatter-add
// One block = 64 edges, 8 waves.  All GEMM math in v_wmma_f32_16x16x32_bf16.
// ---------------------------------------------------------------------------
__global__ __launch_bounds__(256) void edgeconv_mlp_scatter(
    const float*     __restrict__ x,           // [N_NODES][128]
    const long long* __restrict__ edge_index,  // [2][N_EDGES] int64
    const float*     __restrict__ edge_attr,   // [N_EDGES][64]
    const unsigned short* __restrict__ w1t_u,  // bf16 [256][320]
    const float*     __restrict__ b1,          // [256]
    const unsigned short* __restrict__ w2t_u,  // bf16 [128][256]
    const float*     __restrict__ b2,          // [128]
    float*           __restrict__ out)         // [N_NODES][128]
{
  extern __shared__ char smem[];
  __bf16* msg_lds  = (__bf16*)smem;                                   // [64][MS]
  __bf16* h_lds    = (__bf16*)(smem + TILE_E * MS * 2);               // [64][HS]
  int*    rows_lds = (int*)(smem + TILE_E * MS * 2 + TILE_E * HS * 2);
  int*    cols_lds = rows_lds + TILE_E;
  const __bf16* W1T = (const __bf16*)w1t_u;
  const __bf16* W2T = (const __bf16*)w2t_u;

  const int tid  = threadIdx.x;
  const int lane = tid & 31;
  const int wv   = tid >> 5;
  const long long ebase = (long long)blockIdx.x * TILE_E;

  if (tid < TILE_E) {
    rows_lds[tid] = (int)edge_index[ebase + tid];
    cols_lds[tid] = (int)edge_index[(long long)N_EDGES + ebase + tid];
  }
  __syncthreads();

  // ---- Stage message tile: [x[row] | x[col] | edge_attr] -> bf16 in LDS ----
  // 64*320/4 = 5120 float4 chunks; x and out are L2-resident (51MB < 192MB L2)
  for (int i = tid; i < TILE_E * IN_DIM / 4; i += 256) {
    const int g = i << 2;
    const int e = g / IN_DIM;
    const int c = g - e * IN_DIM;
    const float* src;
    if (c < NODE_DIM)          src = x + (size_t)rows_lds[e] * NODE_DIM + c;
    else if (c < 2 * NODE_DIM) src = x + (size_t)cols_lds[e] * NODE_DIM + (c - NODE_DIM);
    else                       src = edge_attr + (size_t)(ebase + e) * EDGE_DIM + (c - 2 * NODE_DIM);
    const float4 f = *(const float4*)src;
    v4bf v;
    v[0] = (__bf16)f.x; v[1] = (__bf16)f.y; v[2] = (__bf16)f.z; v[3] = (__bf16)f.w;
    *(v4bf*)(msg_lds + e * MS + c) = v;                 // ds_store_b64
  }
  __syncthreads();

  // ---- GEMM1: h[64x256] = relu(msg[64x320] @ W1 + b1) ----
  // wave wv owns hidden tiles nt = 2*wv, 2*wv+1; B fragment reused by 4 M-tiles
  for (int t = 0; t < 2; ++t) {
    const int nt = wv * 2 + t;
    v8f acc[4];
    #pragma unroll
    for (int mt = 0; mt < 4; ++mt) acc[mt] = (v8f){0.f,0.f,0.f,0.f,0.f,0.f,0.f,0.f};
    for (int kk = 0; kk < IN_DIM; kk += 32) {
      const v16bf bfr = load_frag_b_glb(W1T, IN_DIM, nt * 16, kk, lane);
      #pragma unroll
      for (int mt = 0; mt < 4; ++mt) {
        const v16bf afr = load_frag_a_lds(msg_lds, MS, mt * 16, kk, lane);
        acc[mt] = WMMA_BF16(afr, bfr, acc[mt]);
      }
    }
    // C layout: VGPR r, lanes 0-15 -> (M=r,      N=lane),
    //                    lanes 16-31 -> (M=r+8, N=lane-16)
    const int   n    = nt * 16 + (lane & 15);
    const float bias = b1[n];
    const int   mb   = (lane >> 4) << 3;
    #pragma unroll
    for (int mt = 0; mt < 4; ++mt) {
      #pragma unroll
      for (int r = 0; r < 8; ++r) {
        float y = acc[mt][r] + bias;
        y = y > 0.f ? y : 0.f;                          // ReLU
        h_lds[(mt * 16 + mb + r) * HS + n] = (__bf16)y;
      }
    }
  }
  __syncthreads();

  // ---- GEMM2: msg_out[64x128] = h[64x256] @ W2 + b2; wave wv owns nt = wv ----
  {
    const int nt = wv;
    v8f acc[4];
    #pragma unroll
    for (int mt = 0; mt < 4; ++mt) acc[mt] = (v8f){0.f,0.f,0.f,0.f,0.f,0.f,0.f,0.f};
    for (int kk = 0; kk < HIDDEN_DIM; kk += 32) {
      const v16bf bfr = load_frag_b_glb(W2T, HIDDEN_DIM, nt * 16, kk, lane);
      #pragma unroll
      for (int mt = 0; mt < 4; ++mt) {
        const v16bf afr = load_frag_a_lds(h_lds, HS, mt * 16, kk, lane);
        acc[mt] = WMMA_BF16(afr, bfr, acc[mt]);
      }
    }
    // Scatter-add per destination node; out is L2-resident so atomics stay on-chip
    const int   n    = nt * 16 + (lane & 15);
    const float bias = b2[n];
    const int   mb   = (lane >> 4) << 3;
    #pragma unroll
    for (int mt = 0; mt < 4; ++mt) {
      #pragma unroll
      for (int r = 0; r < 8; ++r) {
        const int m   = mt * 16 + mb + r;
        const int dst = rows_lds[m];
        atomicAdd(out + (size_t)dst * NODE_DIM + n, acc[mt][r] + bias);
      }
    }
  }
}

// ---------------------------------------------------------------------------
// Launch
// ---------------------------------------------------------------------------
extern "C" void kernel_launch(void* const* d_in, const int* in_sizes, int n_in,
                              void* d_out, int out_size, void* d_ws, size_t ws_size,
                              hipStream_t stream) {
  const float*     x          = (const float*)d_in[0];
  const long long* edge_index = (const long long*)d_in[1];
  const float*     edge_attr  = (const float*)d_in[2];
  const float*     W1         = (const float*)d_in[3];
  const float*     b1         = (const float*)d_in[4];
  const float*     W2         = (const float*)d_in[5];
  const float*     b2         = (const float*)d_in[6];
  float*           out        = (float*)d_out;

  // workspace: bf16 transposed weights (K-contiguous) -- 160KB + 64KB
  unsigned short* w1t = (unsigned short*)d_ws;
  unsigned short* w2t = (unsigned short*)((char*)d_ws + (size_t)IN_DIM * HIDDEN_DIM * 2);

  edgeconv_prep_weights<<<64, 256, 0, stream>>>(W1, W2, w1t, w2t);

  const int n4 = (N_NODES * NODE_DIM) / 4;
  edgeconv_zero_out<<<(n4 + 255) / 256, 256, 0, stream>>>((float4*)out, n4);

  const size_t shmem = (size_t)TILE_E * MS * 2    // msg tile
                     + (size_t)TILE_E * HS * 2    // h tile
                     + (size_t)TILE_E * 2 * 4;    // rows + cols
  edgeconv_mlp_scatter<<<NBLOCKS, 256, shmem, stream>>>(
      x, edge_index, edge_attr, w1t, b1, w2t, b2, out);
}
